// View_652835029217
// MI455X (gfx1250) — matrix operand order repaired and checked by reading, not compiled, not run on hardware
//
#include <hip/hip_runtime.h>

// ---------------------------------------------------------------------------
// NeRF renderer for MI455X (gfx1250, wave32, WMMA).
// H=W=128, S=32 samples, hidden=256 (fixed by setup_inputs).
// ---------------------------------------------------------------------------

typedef __attribute__((ext_vector_type(16))) _Float16 v16h;
typedef __attribute__((ext_vector_type(8)))  float    v8f;

#define HID   256
#define PTS   128      // points per block (4 rays)
#define SMP   32
#define STR   264      // padded LDS stride in halves (528B row -> bank +4/row)

union Frag {
    v16h     h;
    uint4    u[2];
    _Float16 arr[16];
};

// --- pre-kernel: W2 (f32 [K=256][N=256]) -> f16 transposed [N][K] in d_ws ---
__global__ void convert_w2_kernel(const float* __restrict__ W2,
                                  _Float16* __restrict__ w2t) {
    int i = blockIdx.x * 256 + threadIdx.x;   // 65536 total
    int n = i >> 8;
    int k = i & 255;
    w2t[n * 256 + k] = (_Float16)W2[k * 256 + n];
}

static __device__ __forceinline__ v8f wmma_f16(v16h a, v16h b, v8f c) {
    return __builtin_amdgcn_wmma_f32_16x16x32_f16(false, a, false, b,
                                                  (short)0, c, false, false);
}

__global__ __launch_bounds__(256) void nerf_render_kernel(
    const float* __restrict__ Kc,   const float* __restrict__ T,
    const float* __restrict__ mnc,  const float* __restrict__ mxc,
    const float* __restrict__ W1,   const float* __restrict__ b1,
    const _Float16* __restrict__ w2t,
    const float* __restrict__ b2,
    const float* __restrict__ W3,   const float* __restrict__ b3,
    float* __restrict__ out) {

    __shared__ _Float16 h2s[PTS * STR];     // 67584 B: relu(h2) staging
    __shared__ _Float16 w3ts[16 * STR];     //  8448 B: W3^T zero-padded to 16 cols
    __shared__ float    W1s[3 * HID];       //  3072 B
    __shared__ float    b1s[HID];
    __shared__ float    b2s[HID];
    __shared__ float    b3s[4];
    __shared__ float    outS[PTS * 4];      // sigma,r,g,b per point
    __shared__ float    rayL[4];
    __shared__ int      rayHit[4];

    const int tid  = threadIdx.x;
    const int wave = tid >> 5;
    const int lane = tid & 31;
    const int half = lane >> 4;
    const int r    = lane & 15;

    // ---- stage small weights into LDS ----
    for (int i = tid; i < 3 * HID; i += 256) W1s[i] = W1[i];
    for (int i = tid; i < HID; i += 256) { b1s[i] = b1[i]; b2s[i] = b2[i]; }
    if (tid < 4) b3s[tid] = b3[tid];
    for (int i = tid; i < 16 * STR; i += 256) w3ts[i] = (_Float16)0.f;
    __syncthreads();
    for (int i = tid; i < HID * 4; i += 256) {    // W3 is [256][4]
        int k = i >> 2, c = i & 3;
        w3ts[c * STR + k] = (_Float16)W3[i];
    }
    __syncthreads();

    // ---- ray / point geometry (lane owns point pp = wave*16 + r) ----
    const int pp  = wave * 16 + r;
    const int gp  = blockIdx.x * PTS + pp;
    const int ray = gp >> 5;
    const int smp = gp & 31;

    const float fx = Kc[0], fy = Kc[4], cx = Kc[2], cy = Kc[5];
    float u = (float)(ray & 127) + 0.5f;
    float v = (float)(ray >> 7)  + 0.5f;
    float dc0 = (u - cx) / fx, dc1 = (v - cy) / fy, dc2 = 1.0f;
    float inl = rsqrtf(dc0 * dc0 + dc1 * dc1 + 1.0f);
    dc0 *= inl; dc1 *= inl; dc2 = inl;
    float d0 = T[0] * dc0 + T[1] * dc1 + T[2]  * dc2;
    float d1 = T[4] * dc0 + T[5] * dc1 + T[6]  * dc2;
    float d2 = T[8] * dc0 + T[9] * dc1 + T[10] * dc2;
    float o0 = T[3], o1 = T[7], o2 = T[11];
    float mn0 = mnc[0], mn1 = mnc[1], mn2 = mnc[2];
    float mx0 = mxc[0], mx1 = mxc[1], mx2 = mxc[2];

    float i0 = 1.0f / d0, i1 = 1.0f / d1, i2 = 1.0f / d2;
    float a0 = (mn0 - o0) * i0, c0 = (mx0 - o0) * i0;
    float a1 = (mn1 - o1) * i1, c1 = (mx1 - o1) * i1;
    float a2 = (mn2 - o2) * i2, c2 = (mx2 - o2) * i2;
    float tnear = fmaxf(fmaxf(fminf(a0, c0), fminf(a1, c1)), fminf(a2, c2));
    float tfar  = fminf(fminf(fmaxf(a0, c0), fmaxf(a1, c1)), fmaxf(a2, c2));
    tnear = fmaxf(tnear, 0.0f);
    bool hit = tnear < tfar;
    float tfs = hit ? tfar : (tnear + 1.0f);
    float Lr  = tfs - tnear;
    if (lane == 0 && (pp & 31) == 0) {
        rayL[pp >> 5]   = Lr;
        rayHit[pp >> 5] = hit ? 1 : 0;
    }
    float tmid = tnear + Lr * (((float)smp + 0.5f) * (1.0f / 32.0f));
    float nx = (o0 + tmid * d0 - mn0) * (2.0f / (mx0 - mn0)) - 1.0f;
    float ny = (o1 + tmid * d1 - mn1) * (2.0f / (mx1 - mn1)) - 1.0f;
    float nz = (o2 + tmid * d2 - mn2) * (2.0f / (mx2 - mn2)) - 1.0f;

    // ---- layer 1 fused into WMMA A-fragments (per ISA 16-bit A 16x32 layout) ----
    Frag af[8];
#pragma unroll
    for (int kt = 0; kt < 8; ++kt) {
#pragma unroll
        for (int e = 0; e < 16; ++e) {
            int n = kt * 32 + half * 8 + e + ((e >= 8) ? 8 : 0);
            float h = fmaf(nx, W1s[n],
                      fmaf(ny, W1s[HID + n],
                      fmaf(nz, W1s[2 * HID + n], b1s[n])));
            af[kt].arr[e] = (_Float16)fmaxf(h, 0.0f);
        }
    }

    // ---- layer 2: 256x256 via WMMA, B streamed from global f16 W2^T ----
    for (int t = 0; t < 16; t += 2) {
        v8f acc0 = {};
        v8f acc1 = {};
#pragma unroll
        for (int kt = 0; kt < 8; ++kt) {
            int kb = kt * 32 + half * 8;
            const _Float16* p0 = w2t + ((t + 0) * 16 + r) * 256 + kb;
            const _Float16* p1 = w2t + ((t + 1) * 16 + r) * 256 + kb;
            Frag bf0, bf1;
            bf0.u[0] = ((const uint4*)p0)[0];
            bf0.u[1] = ((const uint4*)p0)[2];
            bf1.u[0] = ((const uint4*)p1)[0];
            bf1.u[1] = ((const uint4*)p1)[2];
            acc0 = wmma_f16(af[kt].h, bf0.h, acc0);
            acc1 = wmma_f16(af[kt].h, bf1.h, acc1);
        }
        // bias + relu + stage f16 h2 to LDS (wave-private rows)
#pragma unroll
        for (int q = 0; q < 2; ++q) {
            int n = (t + q) * 16 + r;
            float bb = b2s[n];
#pragma unroll
            for (int j = 0; j < 8; ++j) {
                float hv = fmaxf((q ? acc1[j] : acc0[j]) + bb, 0.0f);
                h2s[(wave * 16 + j + 8 * half) * STR + n] = (_Float16)hv;
            }
        }
    }

    // ---- layer 3: 256 -> 16 (4 live cols) via WMMA ----
    v8f accO = {};
#pragma unroll
    for (int kt = 0; kt < 8; ++kt) {
        const _Float16* ha = h2s + (wave * 16 + r) * STR;
        Frag a, b;
        a.u[0] = *(const uint4*)(ha + kt * 32 + half * 8);
        a.u[1] = *(const uint4*)(ha + kt * 32 + 16 + half * 8);
        const _Float16* wb = w3ts + r * STR + kt * 32 + half * 8;
        b.u[0] = ((const uint4*)wb)[0];
        b.u[1] = ((const uint4*)wb)[2];
        accO = wmma_f16(a.h, b.h, accO);
    }

    // ---- activations: lane column r<4 holds sigma(0)/rgb(1..3) ----
    if (r < 4) {
        float bb = b3s[r];
#pragma unroll
        for (int j = 0; j < 8; ++j) {
            float x = accO[j] + bb;
            float val;
            if (r == 0)
                val = (x > 20.0f) ? x : __logf(1.0f + __expf(x));   // softplus
            else
                val = 1.0f / (1.0f + __expf(-x));                    // sigmoid
            outS[(wave * 16 + j + 8 * half) * 4 + r] = val;
        }
    }
    __syncthreads();

    // ---- per-ray composite: lanes 0..3, one ray each ----
    if (tid < 4) {
        float L2  = rayL[tid];
        int   h   = rayHit[tid];
        float dlt = L2 * (1.0f / 32.0f);
        float csum = 0.0f, cr = 0.0f, cg = 0.0f, cb = 0.0f;
        for (int s = 0; s < SMP; ++s) {
            float4 oS = *(const float4*)&outS[(tid * 32 + s) * 4];
            float dd  = (s == SMP - 1) ? 0.5f * dlt : dlt;
            float si  = oS.x * dd;
            float tex = __expf(-csum);
            float w   = tex * (1.0f - __expf(-si));
            cr += w * oS.y; cg += w * oS.z; cb += w * oS.w;
            csum += si;
        }
        float al = 1.0f - __expf(-csum);
        if (!h) { cr = 0.0f; cg = 0.0f; cb = 0.0f; al = 0.0f; }
        int gray = blockIdx.x * 4 + tid;
        *(float4*)(out + gray * 4) = make_float4(cr, cg, cb, al);
    }
}

extern "C" void kernel_launch(void* const* d_in, const int* in_sizes, int n_in,
                              void* d_out, int out_size, void* d_ws, size_t ws_size,
                              hipStream_t stream) {
    const float* Kc  = (const float*)d_in[0];
    const float* T   = (const float*)d_in[1];
    const float* mnc = (const float*)d_in[2];
    const float* mxc = (const float*)d_in[3];
    const float* W1  = (const float*)d_in[4];
    const float* b1  = (const float*)d_in[5];
    const float* W2  = (const float*)d_in[6];
    const float* b2  = (const float*)d_in[7];
    const float* W3  = (const float*)d_in[8];
    const float* b3  = (const float*)d_in[9];

    _Float16* w2t = (_Float16*)d_ws;   // 256*256 f16 = 128 KB

    convert_w2_kernel<<<256, 256, 0, stream>>>(W2, w2t);

    // 524288 points / 128 per block = 4096 blocks
    nerf_render_kernel<<<4096, 256, 0, stream>>>(
        Kc, T, mnc, mxc, W1, b1, (const _Float16*)w2t, b2, W3, b3,
        (float*)d_out);
}
